// GraphEmbedding_35613868819192
// MI455X (gfx1250) — matrix operand order repaired and checked by reading, not compiled, not run on hardware
//
#include <hip/hip_runtime.h>

// ---------------------------------------------------------------------------
// GNN message passing for MI455X (gfx1250, wave32, WMMA f32<-f16 16x16x32).
// N=100000 nodes, E=800000 edges, HIDDEN=64, 3 layers.
// GEMMs run on v_wmma_f32_16x16x32_f16 with device-side weight pre-swizzle.
// ---------------------------------------------------------------------------

typedef __attribute__((ext_vector_type(16))) _Float16 v16h;
typedef __attribute__((ext_vector_type(8)))  _Float16 v8h;
typedef __attribute__((ext_vector_type(8)))  float    v8f;

union V16 { v16h v; v8h h[2]; };

#define N_NODES 100000
#define N_EDGES 800000
#define HID 64

// ---- workspace layout (bytes). Total ~77.6 MB. ----
#define OFF_X      ((size_t)0)            // N*64 f32  = 25,600,000
#define OFF_X16    ((size_t)25600000)     // N*64 f16  = 12,800,000
#define OFF_AGG    ((size_t)38400000)     // N*64 f32  = 25,600,000 (reused as pre-LN y)
#define OFF_AGG16  ((size_t)64000000)     // N*64 f16  = 12,800,000
#define OFF_INV    ((size_t)76800000)     // N f32     =    400,000
#define OFF_WF     ((size_t)77200000)     // 3*2*8192 f16 = 98,304
#define OFF_PSUM   ((size_t)77298304)     // 512*64 f32 = 131,072
#define OFF_PMAX   ((size_t)77429376)     // 512*64 f32 = 131,072

// ---------------------------------------------------------------------------
__global__ void fill_f32(float* __restrict__ p, float v, int n) {
    for (int i = blockIdx.x * blockDim.x + threadIdx.x; i < n; i += gridDim.x * blockDim.x)
        p[i] = v;
}

__global__ void count_kernel(const int* __restrict__ edges, float* __restrict__ inv) {
    int i = blockIdx.x * blockDim.x + threadIdx.x;
    if (i < N_EDGES) atomicAdd(inv + edges[2 * i + 1], 1.0f);
}

__global__ void invclip_kernel(float* __restrict__ inv) {
    int i = blockIdx.x * blockDim.x + threadIdx.x;
    if (i < N_NODES) inv[i] = 1.0f / fmaxf(inv[i], 1.0f);
}

__global__ void embed_kernel(const int* __restrict__ ti, const float* __restrict__ table,
                             float* __restrict__ x, _Float16* __restrict__ x16) {
    int i = blockIdx.x * blockDim.x + threadIdx.x;
    if (i >= N_NODES * HID) return;
    int node = i >> 6, h = i & 63;
    float v = table[ti[node] * HID + h];
    x[i] = v;
    x16[i] = (_Float16)v;
}

// Pre-swizzle msg_w / upd_w (each 3 x 128 x 64 f32, row-major [K][n]) into
// per-lane WMMA B-fragments: within (layer,mat): frag = kk*4+nt, then
// [lane][16 halves]; lane half h holds W[32*kk + (lane&16)/2 + (h&7) + 2*(h&8)][nt*16 + lane%16].
__global__ void swizzle_kernel(const float* __restrict__ msg_w, const float* __restrict__ upd_w,
                               _Float16* __restrict__ wf) {
    int idx = blockIdx.x * blockDim.x + threadIdx.x;
    if (idx >= 3 * 2 * 16 * 32 * 16) return;
    int h     = idx & 15;
    int lane  = (idx >> 4) & 31;
    int frag  = (idx >> 9) & 15;
    int mat   = (idx >> 13) & 1;
    int layer = idx >> 14;
    int kk = frag >> 2, ntile = frag & 3;
    int K = 32 * kk + ((lane & 16) >> 1) + (h & 7) + ((h & 8) << 1);
    int n = ntile * 16 + (lane & 15);
    const float* W = (mat == 0 ? msg_w : upd_w) + layer * 128 * HID;
    wf[idx] = (_Float16)W[K * HID + n];
}

// 16x16 f32 tile, K=128 split over two f16 source rows (rowA covers K 0..63,
// rowB covers K 64..127). wfrag = swizzled 128x64 weight fragments.
__device__ __forceinline__ v8f gemm_tile(const _Float16* __restrict__ rowA,
                                         const _Float16* __restrict__ rowB,
                                         const _Float16* __restrict__ wfrag,
                                         int lane, int nt) {
    v8f acc = {};
    const int hi = (lane & 16) >> 1;  // 0 for lanes 0-15, 8 for lanes 16-31
#pragma unroll
    for (int kk = 0; kk < 4; ++kk) {
        const _Float16* rp = (kk < 2) ? rowA : rowB;
        int foff = ((kk & 1) << 5) + hi;
        V16 a;
        a.h[0] = *(const v8h*)(rp + foff);        // K local 0..7  (or 8..15)
        a.h[1] = *(const v8h*)(rp + foff + 16);   // K local 16..23 (or 24..31)
        v16h b = *(const v16h*)(wfrag + (((kk << 2) + nt) * 32 + lane) * 16);
        acc = __builtin_amdgcn_wmma_f32_16x16x32_f16(false, a.v, false, b,
                                                     (short)0, acc, false, false);
    }
    return acc;
}

// One wave = one (16-edge tile, 16-column tile). msg = relu(A@W+b), atomically
// scattered into agg[tgt]. ReLU zeros skip the atomic entirely.
__global__ __launch_bounds__(256) void msg_kernel(const int* __restrict__ edges,
                                                  const _Float16* __restrict__ x16,
                                                  const _Float16* __restrict__ wfrag,
                                                  const float* __restrict__ bias,
                                                  float* __restrict__ agg) {
    int wave = blockIdx.x * (blockDim.x >> 5) + (threadIdx.x >> 5);
    int lane = threadIdx.x & 31;
    int tile = wave >> 2;
    int nt   = wave & 3;
    int m = lane & 15;
    int e = tile * 16 + m;
    int s = edges[2 * e];
    int t = edges[2 * e + 1];
    v8f acc = gemm_tile(x16 + (size_t)s * HID, x16 + (size_t)t * HID, wfrag, lane, nt);
    int n = nt * 16 + m;
    float bv = bias[n];
    int rbase = (lane & 16) >> 1;  // C rows 0..7 (lanes<16) or 8..15
#pragma unroll
    for (int r = 0; r < 8; ++r) {
        float v = acc[r] + bv;
        if (v > 0.0f) {
            int tr = edges[2 * (tile * 16 + rbase + r) + 1];
            atomicAdd(agg + (size_t)tr * HID + n, v);
        }
    }
}

__global__ void norm_agg_kernel(const float* __restrict__ agg, const float* __restrict__ inv,
                                _Float16* __restrict__ agg16) {
    int i = blockIdx.x * blockDim.x + threadIdx.x;
    if (i >= N_NODES * HID) return;
    agg16[i] = (_Float16)(agg[i] * inv[i >> 6]);
}

// One wave = one (16-node tile, 16-column tile). y = relu(A@W+b) + x (pre-LN).
__global__ __launch_bounds__(256) void upd_kernel(const _Float16* __restrict__ x16,
                                                  const _Float16* __restrict__ agg16,
                                                  const float* __restrict__ x,
                                                  const _Float16* __restrict__ wfrag,
                                                  const float* __restrict__ bias,
                                                  float* __restrict__ y) {
    int wave = blockIdx.x * (blockDim.x >> 5) + (threadIdx.x >> 5);
    int lane = threadIdx.x & 31;
    int tile = wave >> 2;
    int nt   = wave & 3;
    int m = lane & 15;
    size_t node = (size_t)(tile * 16 + m);
    v8f acc = gemm_tile(x16 + node * HID, agg16 + node * HID, wfrag, lane, nt);
    int n = nt * 16 + m;
    float bv = bias[n];
    int rbase = (lane & 16) >> 1;
#pragma unroll
    for (int r = 0; r < 8; ++r) {
        size_t row = (size_t)(tile * 16 + rbase + r);
        float v = acc[r] + bv;
        v = v > 0.0f ? v : 0.0f;
        y[row * HID + n] = v + x[row * HID + n];
    }
}

// One wave per node: LayerNorm over 64 features (2 per lane), write f32 + f16.
__global__ __launch_bounds__(256) void ln_kernel(const float* __restrict__ y,
                                                 const float* __restrict__ g,
                                                 const float* __restrict__ b,
                                                 float* __restrict__ x,
                                                 _Float16* __restrict__ x16) {
    int wave = blockIdx.x * (blockDim.x >> 5) + (threadIdx.x >> 5);
    int lane = threadIdx.x & 31;
    if (wave >= N_NODES) return;
    const float* row = y + (size_t)wave * HID;
    float v0 = row[lane], v1 = row[lane + 32];
    float s = v0 + v1;
#pragma unroll
    for (int off = 16; off; off >>= 1) s += __shfl_xor(s, off, 32);
    float mu = s * (1.0f / 64.0f);
    float d0 = v0 - mu, d1 = v1 - mu;
    float q = d0 * d0 + d1 * d1;
#pragma unroll
    for (int off = 16; off; off >>= 1) q += __shfl_xor(q, off, 32);
    float rs = rsqrtf(q * (1.0f / 64.0f) + 1e-5f);
    float o0 = d0 * rs * g[lane] + b[lane];
    float o1 = d1 * rs * g[lane + 32] + b[lane + 32];
    size_t base = (size_t)wave * HID;
    x[base + lane] = o0;
    x[base + lane + 32] = o1;
    x16[base + lane] = (_Float16)o0;
    x16[base + lane + 32] = (_Float16)o1;
}

__global__ void reduce_partial(const float* __restrict__ x, float* __restrict__ psum,
                               float* __restrict__ pmax) {
    int t = threadIdx.x, bnum = blockIdx.x;
    float s = 0.0f, mx = -3.402823466e38f;
    for (int n = bnum; n < N_NODES; n += gridDim.x) {
        float v = x[(size_t)n * HID + t];
        s += v;
        mx = fmaxf(mx, v);
    }
    psum[bnum * HID + t] = s;
    pmax[bnum * HID + t] = mx;
}

__global__ void reduce_final(const float* __restrict__ psum, const float* __restrict__ pmax,
                             float* __restrict__ out, int nb) {
    int t = threadIdx.x;
    float s = 0.0f, mx = -3.402823466e38f;
    for (int bb = 0; bb < nb; ++bb) {
        s += psum[bb * HID + t];
        mx = fmaxf(mx, pmax[bb * HID + t]);
    }
    out[t] = s * (1.0f / (float)N_NODES);
    out[HID + t] = mx;
}

// ---------------------------------------------------------------------------
extern "C" void kernel_launch(void* const* d_in, const int* in_sizes, int n_in,
                              void* d_out, int out_size, void* d_ws, size_t ws_size,
                              hipStream_t stream) {
    const int*   type_indices = (const int*)d_in[0];
    const int*   edges        = (const int*)d_in[1];
    const float* embed_table  = (const float*)d_in[2];
    const float* msg_w        = (const float*)d_in[3];
    const float* msg_b        = (const float*)d_in[4];
    const float* upd_w        = (const float*)d_in[5];
    const float* upd_b        = (const float*)d_in[6];
    const float* ln_g         = (const float*)d_in[7];
    const float* ln_b         = (const float*)d_in[8];
    (void)in_sizes; (void)n_in; (void)out_size; (void)ws_size;

    char* ws = (char*)d_ws;
    float*    x     = (float*)(ws + OFF_X);
    _Float16* x16   = (_Float16*)(ws + OFF_X16);
    float*    agg   = (float*)(ws + OFF_AGG);     // also pre-LN y buffer
    _Float16* agg16 = (_Float16*)(ws + OFF_AGG16);
    float*    inv   = (float*)(ws + OFF_INV);
    _Float16* wf    = (_Float16*)(ws + OFF_WF);
    float*    psum  = (float*)(ws + OFF_PSUM);
    float*    pmax  = (float*)(ws + OFF_PMAX);

    // degree -> 1/max(cnt,1)
    fill_f32<<<512, 256, 0, stream>>>(inv, 0.0f, N_NODES);
    count_kernel<<<(N_EDGES + 255) / 256, 256, 0, stream>>>(edges, inv);
    invclip_kernel<<<(N_NODES + 255) / 256, 256, 0, stream>>>(inv);

    // x = embed_table[type_indices], plus f16 shadow
    embed_kernel<<<(N_NODES * HID + 255) / 256, 256, 0, stream>>>(type_indices, embed_table, x, x16);

    // swizzle all weights into WMMA B-fragment order
    swizzle_kernel<<<(3 * 2 * 16 * 32 * 16 + 255) / 256, 256, 0, stream>>>(msg_w, upd_w, wf);

    for (int L = 0; L < 3; ++L) {
        const _Float16* wmsg = wf + (size_t)(L * 2 + 0) * 8192;
        const _Float16* wupd = wf + (size_t)(L * 2 + 1) * 8192;

        fill_f32<<<2048, 256, 0, stream>>>(agg, 0.0f, N_NODES * HID);

        // edge GEMM + relu + scatter-add: 50000 edge tiles * 4 ntiles = 200000 waves
        msg_kernel<<<25000, 256, 0, stream>>>(edges, x16, wmsg, msg_b + L * HID, agg);

        // agg / cnt -> f16
        norm_agg_kernel<<<(N_NODES * HID + 255) / 256, 256, 0, stream>>>(agg, inv, agg16);

        // node GEMM + relu + residual: 6250 node tiles * 4 ntiles = 25000 waves
        upd_kernel<<<3125, 256, 0, stream>>>(x16, agg16, x, wupd, upd_b + L * HID, agg);

        // LayerNorm: one wave per node
        ln_kernel<<<12500, 256, 0, stream>>>(agg, ln_g + L * HID, ln_b + L * HID, x, x16);
    }

    reduce_partial<<<512, HID, 0, stream>>>(x, psum, pmax);
    reduce_final<<<1, HID, 0, stream>>>(psum, pmax, (float*)d_out, 512);
}
